// MultiAttention_8718783611483
// MI455X (gfx1250) — matrix operand order repaired and checked by reading, not compiled, not run on hardware
//
#include <hip/hip_runtime.h>
#include <stdint.h>

// MultiHeadAttention for MI455X (gfx1250, wave32, WMMA bf16).
// N=4096, DIM=128, HEAD=8. Workspace use ~36 MB (bf16 staging + projections).

#define DIMC   128
#define HEADS  8
#define HD     1024   // HEADS*DIMC

typedef __bf16 v16bf __attribute__((ext_vector_type(16)));
typedef float  v8f   __attribute__((ext_vector_type(8)));

// Types matching the async-LDS builtin's prototype: int4 in AS(1)/AS(3).
typedef int v4i_vs __attribute__((vector_size(16)));
typedef v4i_vs __attribute__((address_space(1))) v4i_g;
typedef v4i_vs __attribute__((address_space(3))) v4i_l;

union Frag {
  v16bf v;
  uint4 u[2];
  unsigned short h[16];
};

__device__ __forceinline__ unsigned short f2bf(float x) {
  unsigned int u = __float_as_uint(x);
  u += 0x7FFFu + ((u >> 16) & 1u);      // round-to-nearest-even
  return (unsigned short)(u >> 16);
}

__device__ __forceinline__ v8f vzero8() {
  v8f x;
#pragma unroll
  for (int i = 0; i < 8; ++i) x[i] = 0.0f;
  return x;
}

// ---- CDNA5 async memory->LDS copy (ASYNCcnt-tracked, bypasses VGPRs) ------
__device__ __forceinline__ void async_ld_b128(const unsigned short* g,
                                              unsigned short* l) {
#if __has_builtin(__builtin_amdgcn_global_load_async_to_lds_b128)
  __builtin_amdgcn_global_load_async_to_lds_b128(
      (v4i_g*)(uintptr_t)g,      // global src (AS1 int4*)
      (v4i_l*)(uintptr_t)l,      // LDS dst: low 32 bits = allocation offset
      0, 0);
#else
  unsigned loff = (unsigned)(uintptr_t)l;
  asm volatile("global_load_async_to_lds_b128 %0, %1, off"
               :: "v"(loff), "v"(g) : "memory");
#endif
}

__device__ __forceinline__ void wait_async0() {
#if __has_builtin(__builtin_amdgcn_s_wait_asynccnt)
  __builtin_amdgcn_s_wait_asynccnt(0);
#else
  asm volatile("s_wait_asynccnt 0x0" ::: "memory");
#endif
}

// ---------------------------------------------------------------- converters
__global__ void cvt_bf16_kernel(const float* __restrict__ s,
                                unsigned short* __restrict__ d, int n) {
  int i = blockIdx.x * blockDim.x + threadIdx.x;
  if (i < n) d[i] = f2bf(s[i]);
}

// src [R][C] fp32 -> dst [C][R] bf16 (transpose)
__global__ void cvt_tr_bf16_kernel(const float* __restrict__ s,
                                   unsigned short* __restrict__ d, int R, int C) {
  int i = blockIdx.x * blockDim.x + threadIdx.x;
  if (i < R * C) {
    int r = i / C, c = i % C;
    d[c * R + r] = f2bf(s[i]);
  }
}

// ------------------------------------------------------- projection GEMM
// P[N][1024] = bf16( X[N][128] @ W[128][1024] + bias ).  Wt = W^T stored [1024][128].
// One wave -> 16 rows x 64 cols (A fragment reused across 4 B tiles).
__global__ void __launch_bounds__(256) proj_gemm_kernel(
    const unsigned short* __restrict__ X, const unsigned short* __restrict__ Wt,
    const float* __restrict__ bias, unsigned short* __restrict__ P) {
  const int lane = threadIdx.x & 31;
  const int wave = threadIdx.x >> 5;
  const int wg   = blockIdx.x * 8 + wave;
  const int row0 = (wg >> 4) * 16;
  const int col0 = (wg & 15) * 64;
  const int n  = lane & 15;
  const int hi = lane >> 4;

  v8f acc[4];
#pragma unroll
  for (int c = 0; c < 4; ++c) acc[c] = vzero8();

#pragma unroll
  for (int ks = 0; ks < 4; ++ks) {               // K = 128 = 4 x 32
    Frag a;
    const unsigned short* ap = X + (row0 + n) * DIMC + ks * 32 + hi * 8;
    a.u[0] = *(const uint4*)ap;
    a.u[1] = *(const uint4*)(ap + 16);
#pragma unroll
    for (int c = 0; c < 4; ++c) {
      Frag b;
      const unsigned short* bp = Wt + (col0 + c * 16 + n) * DIMC + ks * 32 + hi * 8;
      b.u[0] = *(const uint4*)bp;
      b.u[1] = *(const uint4*)(bp + 16);
      acc[c] = __builtin_amdgcn_wmma_f32_16x16x32_bf16(
          false, a.v, false, b.v, (short)0, acc[c], false, false);
    }
  }
#pragma unroll
  for (int c = 0; c < 4; ++c) {
    const int col = col0 + c * 16 + n;
    const float bv = bias[col];
#pragma unroll
    for (int i = 0; i < 8; ++i) {
      const int r = row0 + i + hi * 8;
      P[r * HD + col] = f2bf(acc[c][i] + bv);
    }
  }
}

// ------------------------------------------------------------ flash attention
// Q,K,V: [N][1024] bf16 (head h at cols h*128). Ctx out: [N][1024] bf16.
// Block = (head, 128 query rows); 8 waves x 16 rows. One pass, online softmax.
// S^T = K.Q^T so softmax dim sits on the VGPR axis; ctx^T = V^T.P from LDS.
// K tile staged via GLOBAL_LOAD_ASYNC_TO_LDS_B128 (ASYNCcnt).
__global__ void __launch_bounds__(256) attn_kernel(
    const unsigned short* __restrict__ Q, const unsigned short* __restrict__ K,
    const unsigned short* __restrict__ V, unsigned short* __restrict__ Ctx, int N) {
  __shared__ unsigned short Ks[32 * 136];   // pitch 136 halves: 16B-aligned, conflict-free
  __shared__ unsigned short Vt[128 * 40];   // V^T tile, pitch 40 halves

  const int qblocks = N >> 7;
  const int h  = blockIdx.x / qblocks;
  const int qb = blockIdx.x % qblocks;
  const int lane = threadIdx.x & 31;
  const int wave = threadIdx.x >> 5;
  const int q0 = qb * 128 + wave * 16;
  const int n  = lane & 15;
  const int hi = lane >> 4;
  const float SCALE = 0.08838834764831845f;   // 1/sqrt(128)

  // Q as B-matrix fragments: lane holds query col n, k = d in table order (8h+s, 16+8h+s-8)
  Frag qf[4];
#pragma unroll
  for (int d = 0; d < 4; ++d) {
    const unsigned short* qp = Q + (q0 + n) * HD + h * DIMC + d * 32 + hi * 8;
    qf[d].u[0] = *(const uint4*)qp;
    qf[d].u[1] = *(const uint4*)(qp + 16);
  }

  v8f out[8];                                  // ctx^T tiles, rows = d
#pragma unroll
  for (int t = 0; t < 8; ++t) out[t] = vzero8();
  float mrun = -1e30f, lrun = 0.0f;

  const int r  = threadIdx.x >> 3;             // staging row 0..31
  const int c0 = (threadIdx.x & 7) * 16;       // staging col chunk

  for (int m0 = 0; m0 < N; m0 += 32) {
    __syncthreads();
    { // stage K tile via async DMA-to-LDS; V tile transposed through VGPRs
      const unsigned short* kg = K + (m0 + r) * HD + h * DIMC + c0;
      unsigned short* kl = &Ks[r * 136 + c0];
      async_ld_b128(kg, kl);
      async_ld_b128(kg + 8, kl + 8);

      union { uint4 u[2]; unsigned short s[16]; } vv;
      const uint4* vsrc = (const uint4*)(V + (m0 + r) * HD + h * DIMC + c0);
      vv.u[0] = vsrc[0];
      vv.u[1] = vsrc[1];
#pragma unroll
      for (int j = 0; j < 16; ++j) Vt[(c0 + j) * 40 + r] = vv.s[j];

      if (m0 + 32 < N) {   // pull next tiles toward the WGP while we compute
        __builtin_prefetch(K + (m0 + 32 + r) * HD + h * DIMC + c0, 0, 3);
        __builtin_prefetch(V + (m0 + 32 + r) * HD + h * DIMC + c0, 0, 3);
      }
    }
    wait_async0();          // our async K chunks are in LDS
    __syncthreads();        // everyone's staging visible

    // S^T tiles: m rows (keys) x n cols (queries), K-dim = d = 128
    v8f s0 = vzero8(), s1 = vzero8();
#pragma unroll
    for (int d = 0; d < 4; ++d) {
      Frag a0, a1;
      const unsigned short* k0p = &Ks[(n)      * 136 + d * 32 + hi * 8];
      const unsigned short* k1p = &Ks[(16 + n) * 136 + d * 32 + hi * 8];
      a0.u[0] = *(const uint4*)k0p; a0.u[1] = *(const uint4*)(k0p + 16);
      a1.u[0] = *(const uint4*)k1p; a1.u[1] = *(const uint4*)(k1p + 16);
      s0 = __builtin_amdgcn_wmma_f32_16x16x32_bf16(false, a0.v, false, qf[d].v,
                                                   (short)0, s0, false, false);
      s1 = __builtin_amdgcn_wmma_f32_16x16x32_bf16(false, a1.v, false, qf[d].v,
                                                   (short)0, s1, false, false);
    }

    // online softmax along m (8 in-lane + xor-16 cross-half)
    float bm = fmaxf(s0[0], s1[0]);
#pragma unroll
    for (int i = 1; i < 8; ++i) bm = fmaxf(bm, fmaxf(s0[i], s1[i]));
    bm = fmaxf(bm, __shfl_xor(bm, 16, 32));
    const float mnew  = fmaxf(mrun, bm * SCALE);
    const float alpha = __expf(mrun - mnew);
    float p0[8], p1[8], rs = 0.0f;
#pragma unroll
    for (int i = 0; i < 8; ++i) {
      p0[i] = __expf(s0[i] * SCALE - mnew);
      p1[i] = __expf(s1[i] * SCALE - mnew);
      rs += p0[i] + p1[i];
    }
    rs += __shfl_xor(rs, 16, 32);
    lrun = lrun * alpha + rs;
    mrun = mnew;

#pragma unroll
    for (int t = 0; t < 8; ++t)
#pragma unroll
      for (int i = 0; i < 8; ++i) out[t][i] *= alpha;

    // C-layout P (two 16x16 tiles) -> B-layout 32x16 bf16 via xor-16 exchange
    Frag bp_;
#pragma unroll
    for (int i = 0; i < 8; ++i) {
      const float y0 = __shfl_xor(p0[i], 16, 32);
      const float y1 = __shfl_xor(p1[i], 16, 32);
      bp_.h[i]     = f2bf(hi ? y1 : p0[i]);
      bp_.h[i + 8] = f2bf(hi ? p1[i] : y0);
    }

    // ctx^T += V^T (A, from LDS) x P (B)
#pragma unroll
    for (int dt = 0; dt < 8; ++dt) {
      Frag a;
      const unsigned short* vp = &Vt[(dt * 16 + n) * 40 + hi * 16];
      a.u[0] = *(const uint4*)vp;
      a.u[1] = *(const uint4*)(vp + 8);
      out[dt] = __builtin_amdgcn_wmma_f32_16x16x32_bf16(
          false, a.v, false, bp_.v, (short)0, out[dt], false, false);
    }
  }

  // epilogue: divide by l, transpose-free store (rows i are contiguous d)
  const float inv = 1.0f / lrun;
#pragma unroll
  for (int dt = 0; dt < 8; ++dt) {
    union { uint4 u; unsigned short s[8]; } pk;
#pragma unroll
    for (int i = 0; i < 8; ++i) pk.s[i] = f2bf(out[dt][i] * inv);
    unsigned short* dst = Ctx + (q0 + n) * HD + h * DIMC + dt * 16 + hi * 8;
    *(uint4*)dst = pk.u;
  }
}

// ------------------------------------------------------------- output GEMM
// Out[N][128] f32 = Ctx[N][1024] @ Wo[1024][128] + bo.  Wot = Wo^T [128][1024].
__global__ void __launch_bounds__(256) out_gemm_kernel(
    const unsigned short* __restrict__ Ctx, const unsigned short* __restrict__ Wot,
    const float* __restrict__ bo, float* __restrict__ Out) {
  const int lane = threadIdx.x & 31;
  const int wave = threadIdx.x >> 5;
  const int wg   = blockIdx.x * 8 + wave;
  const int row0 = (wg >> 1) * 16;
  const int col0 = (wg & 1) * 64;
  const int n  = lane & 15;
  const int hi = lane >> 4;

  v8f acc[4];
#pragma unroll
  for (int c = 0; c < 4; ++c) acc[c] = vzero8();

  for (int ks = 0; ks < 32; ++ks) {            // K = 1024
    Frag a;
    const unsigned short* ap = Ctx + (row0 + n) * HD + ks * 32 + hi * 8;
    a.u[0] = *(const uint4*)ap;
    a.u[1] = *(const uint4*)(ap + 16);
#pragma unroll
    for (int c = 0; c < 4; ++c) {
      Frag b;
      const unsigned short* bp = Wot + (col0 + c * 16 + n) * HD + ks * 32 + hi * 8;
      b.u[0] = *(const uint4*)bp;
      b.u[1] = *(const uint4*)(bp + 16);
      acc[c] = __builtin_amdgcn_wmma_f32_16x16x32_bf16(
          false, a.v, false, b.v, (short)0, acc[c], false, false);
    }
  }
#pragma unroll
  for (int c = 0; c < 4; ++c) {
    const int col = col0 + c * 16 + n;
    const float bv = bo[col];
#pragma unroll
    for (int i = 0; i < 8; ++i) {
      Out[(row0 + i + hi * 8) * DIMC + col] = acc[c][i] + bv;
    }
  }
}

// ------------------------------------------------------------------ launcher
extern "C" void kernel_launch(void* const* d_in, const int* in_sizes, int n_in,
                              void* d_out, int out_size, void* d_ws, size_t ws_size,
                              hipStream_t stream) {
  const float* q  = (const float*)d_in[0];
  const float* k  = (const float*)d_in[1];
  const float* v  = (const float*)d_in[2];
  const float* Wq = (const float*)d_in[3];
  const float* bq = (const float*)d_in[4];
  const float* Wk = (const float*)d_in[5];
  const float* bk = (const float*)d_in[6];
  const float* Wv = (const float*)d_in[7];
  const float* bv = (const float*)d_in[8];
  const float* Wo = (const float*)d_in[9];
  const float* bo = (const float*)d_in[10];
  float* out = (float*)d_out;
  const int N = in_sizes[0] / DIMC;            // 4096

  uint8_t* ws = (uint8_t*)d_ws;
  size_t off = 0;
  auto alloc = [&](size_t bytes) -> uint8_t* {
    uint8_t* p = ws + off;
    off += (bytes + 255) & ~(size_t)255;
    return p;
  };
  unsigned short* qb_ = (unsigned short*)alloc((size_t)N * DIMC * 2);
  unsigned short* kb_ = (unsigned short*)alloc((size_t)N * DIMC * 2);
  unsigned short* vb_ = (unsigned short*)alloc((size_t)N * DIMC * 2);
  unsigned short* Wqt = (unsigned short*)alloc((size_t)DIMC * HD * 2);
  unsigned short* Wkt = (unsigned short*)alloc((size_t)DIMC * HD * 2);
  unsigned short* Wvt = (unsigned short*)alloc((size_t)DIMC * HD * 2);
  unsigned short* Wot = (unsigned short*)alloc((size_t)HD * DIMC * 2);
  unsigned short* Qp  = (unsigned short*)alloc((size_t)N * HD * 2);
  unsigned short* Kp  = (unsigned short*)alloc((size_t)N * HD * 2);
  unsigned short* Vp  = (unsigned short*)alloc((size_t)N * HD * 2);
  unsigned short* Ctx = (unsigned short*)alloc((size_t)N * HD * 2);
  // total ~36.4 MB

  const int nx = N * DIMC;
  cvt_bf16_kernel<<<(nx + 255) / 256, 256, 0, stream>>>(q, qb_, nx);
  cvt_bf16_kernel<<<(nx + 255) / 256, 256, 0, stream>>>(k, kb_, nx);
  cvt_bf16_kernel<<<(nx + 255) / 256, 256, 0, stream>>>(v, vb_, nx);
  const int nw = DIMC * HD;
  cvt_tr_bf16_kernel<<<(nw + 255) / 256, 256, 0, stream>>>(Wq, Wqt, DIMC, HD);
  cvt_tr_bf16_kernel<<<(nw + 255) / 256, 256, 0, stream>>>(Wk, Wkt, DIMC, HD);
  cvt_tr_bf16_kernel<<<(nw + 255) / 256, 256, 0, stream>>>(Wv, Wvt, DIMC, HD);
  cvt_tr_bf16_kernel<<<(nw + 255) / 256, 256, 0, stream>>>(Wo, Wot, HD, DIMC);

  const int projBlocks = (N / 16) * 16 / 8;    // 512
  proj_gemm_kernel<<<projBlocks, 256, 0, stream>>>(qb_, Wqt, bq, Qp);
  proj_gemm_kernel<<<projBlocks, 256, 0, stream>>>(kb_, Wkt, bk, Kp);
  proj_gemm_kernel<<<projBlocks, 256, 0, stream>>>(vb_, Wvt, bv, Vp);

  attn_kernel<<<HEADS * (N / 128), 256, 0, stream>>>(Qp, Kp, Vp, Ctx, N);

  out_gemm_kernel<<<(N / 16) * 2 / 8, 256, 0, stream>>>(Ctx, Wot, bo, out);
}
